// EGNN_dynamics_AD2_cat_14508399525997
// MI455X (gfx1250) — compile-verified, matrix operand order checked by reading, not run on hardware
//
#include <hip/hip_runtime.h>
#include <hip/hip_bf16.h>

typedef __attribute__((ext_vector_type(16))) _Float16 v16h;
typedef __attribute__((ext_vector_type(4)))  _Float16 v4h;
typedef __attribute__((ext_vector_type(8)))  float    v8f;
typedef __attribute__((ext_vector_type(4)))  float    v4f;
typedef __attribute__((ext_vector_type(4)))  unsigned int v4u;

#define NPART 22
#define HDIM  64
#define EPB   462   // directed edges per graph
#define EPAD  464   // padded to 29 tiles of 16
#define NTILE 29
#define NLAY  5

// workspace offsets (in halves)
#define OFF_WE1T 0        // [5][64][160]
#define OFF_WE2T 51200    // [5][64][64]
#define OFF_WC1T 71680    // [5][64][64]
#define OFF_WN1T 92160    // [5][64][128]
#define OFF_WN2T 133120   // [5][64][64]
#define WS_HALVES 153600

union H16 { v16h h; v4u u[2]; };

__device__ __forceinline__ v8f wmma16(v16h a, v16h b, v8f c) {
  return __builtin_amdgcn_wmma_f32_16x16x32_f16(false, a, false, b, (short)0, c,
                                                false, false);
}

// A-fragment (16x32 f16): lane&15 = row, (lane>>4)*8 = K subgroup base;
// elements 0..7 = K kg..kg+7, elements 8..15 = K 16+kg..16+kg+7.
__device__ __forceinline__ v16h frag_lds(const _Float16* base, int strideH, int kt, int lane) {
  int mn = lane & 15;
  int kg = (lane >> 4) * 8;
  const _Float16* p = base + mn * strideH + kt * 32;
  H16 r;
  r.u[0] = *(const v4u*)(p + kg);
  r.u[1] = *(const v4u*)(p + 16 + kg);
  return r.h;
}

// B-fragment (32x16 f16) from global weight stored [N][K] (K padded mult of 32).
__device__ __forceinline__ v16h frag_glb(const _Float16* base, int Kp, int nt, int kt, int lane) {
  int n = nt * 16 + (lane & 15);
  int kg = (lane >> 4) * 8;
  const _Float16* p = base + n * Kp + kt * 32;
  H16 r;
  r.u[0] = *(const v4u*)(p + kg);
  r.u[1] = *(const v4u*)(p + 16 + kg);
  return r.h;
}

// silu via v_rcp_f32 (1-ulp) instead of the IEEE div macro
__device__ __forceinline__ float siluf(float x) {
  return x * __builtin_amdgcn_rcpf(1.0f + __expf(-x));
}
__device__ __forceinline__ float sigmf(float x) {
  return __builtin_amdgcn_rcpf(1.0f + __expf(-x));
}

// ---------------- weight prep: f32 [K][N] -> f16 [N][Kpad] ----------------
__global__ void prep_weights(const float* __restrict__ We1, const float* __restrict__ We2,
                             const float* __restrict__ Wc1, const float* __restrict__ Wn1,
                             const float* __restrict__ Wn2, _Float16* __restrict__ ws) {
  int i = blockIdx.x * 256 + threadIdx.x;
  if (i < 51200) {                       // We1T [5][64][160] <- We1 [5][130][64]
    int k = i % 160, n = (i / 160) % 64, l = i / 10240;
    ws[OFF_WE1T + i] = (k < 130) ? (_Float16)We1[(l * 130 + k) * 64 + n] : (_Float16)0.0f;
    return;
  }
  i -= 51200;
  if (i < 20480) {                       // We2T [5][64][64]
    int k = i & 63, n = (i >> 6) & 63, l = i >> 12;
    ws[OFF_WE2T + i] = (_Float16)We2[(l * 64 + k) * 64 + n]; return;
  }
  i -= 20480;
  if (i < 20480) {                       // Wc1T [5][64][64]
    int k = i & 63, n = (i >> 6) & 63, l = i >> 12;
    ws[OFF_WC1T + i] = (_Float16)Wc1[(l * 64 + k) * 64 + n]; return;
  }
  i -= 20480;
  if (i < 40960) {                       // Wn1T [5][64][128]
    int k = i & 127, n = (i >> 7) & 63, l = i >> 13;
    ws[OFF_WN1T + i] = (_Float16)Wn1[(l * 128 + k) * 64 + n]; return;
  }
  i -= 40960;
  if (i < 20480) {                       // Wn2T [5][64][64]
    int k = i & 63, n = (i >> 6) & 63, l = i >> 12;
    ws[OFF_WN2T + i] = (_Float16)Wn2[(l * 64 + k) * 64 + n];
  }
}

// ---------------- fused EGNN forward: one workgroup per graph ----------------
__global__ __launch_bounds__(256) void egnn_fused(
    const float* __restrict__ t, const float* __restrict__ x,
    const float* __restrict__ h_init, const int* __restrict__ rows,
    const int* __restrict__ cols, const float* __restrict__ Wemb,
    const float* __restrict__ bemb, const float* __restrict__ be1,
    const float* __restrict__ be2, const float* __restrict__ Watt,
    const float* __restrict__ batt, const float* __restrict__ bn1,
    const float* __restrict__ bn2, const float* __restrict__ bc1,
    const float* __restrict__ Wc2, const _Float16* __restrict__ wsW,
    float* __restrict__ out)
{
  const int b = blockIdx.x;
  const int tid = threadIdx.x;
  const int lane = tid & 31;
  const int wave = tid >> 5;

  __shared__ __align__(16) float hsh[NPART * HDIM];   // node features
  __shared__ __align__(16) float agg[NPART * HDIM];   // message segment-sum
  __shared__ float crd[NPART * 4];
  __shared__ float crd0[NPART * 4];
  __shared__ float cacc[NPART * 4];                   // coord-update segment-sum
  __shared__ float eattr[EPAD];
  __shared__ short er[EPAD], ec[EPAD];
  __shared__ __align__(16) _Float16 Ast[8][16 * 168]; // per-wave f16 A stage
  __shared__ __align__(16) float    Fst[8][16 * 68];  // per-wave f32 stage
  __shared__ float meanv[3];

  // ---- init: embedding, coords, local edge table ----
  const float tb = t[b];
  for (int idx = tid; idx < NPART * HDIM; idx += 256) {
    int p = idx >> 6, n = idx & 63;
    float a = bemb[n] + tb * Wemb[22 * 64 + n];
#pragma unroll
    for (int k = 0; k < NPART; ++k) a += h_init[p * NPART + k] * Wemb[k * 64 + n];
    hsh[idx] = a;
    agg[idx] = 0.0f;
  }
  for (int idx = tid; idx < NPART * 3; idx += 256) {
    int p = idx / 3, d = idx - p * 3;
    float v = x[b * (NPART * 3) + idx];
    crd[p * 4 + d] = v; crd0[p * 4 + d] = v; cacc[p * 4 + d] = 0.0f;
  }
  for (int e = tid; e < EPAD; e += 256) {
    int r = 0, c = 0;
    if (e < EPB) { r = rows[b * EPB + e] - b * NPART; c = cols[b * EPB + e] - b * NPART; }
    er[e] = (short)r; ec[e] = (short)c;
  }
  __syncthreads();
  for (int e = tid; e < EPAD; e += 256) {             // fixed edge_attr @ initial coords
    float s = 0.0f;
    if (e < EPB) {
      int r = er[e], c = ec[e];
#pragma unroll
      for (int d = 0; d < 3; ++d) { float df = crd[r*4+d] - crd[c*4+d]; s += df * df; }
    }
    eattr[e] = s;
  }
  __syncthreads();

  _Float16* A = Ast[wave];
  float*    F = Fst[wave];
  const int ncol = lane & 15;
  const int mrow = (lane >> 4) * 8;
  const v4f vz4 = {};

  for (int l = 0; l < NLAY; ++l) {
    const _Float16* We1T = wsW + OFF_WE1T + l * 10240;
    const _Float16* We2T = wsW + OFF_WE2T + l * 4096;
    const _Float16* Wc1T = wsW + OFF_WC1T + l * 4096;
    const _Float16* Wn1T = wsW + OFF_WN1T + l * 8192;
    const _Float16* Wn2T = wsW + OFF_WN2T + l * 4096;

    // ================= edge phase: waves stride over 16-edge tiles =================
    for (int tile = wave; tile < NTILE; tile += 8) {
      const int ebase = tile * 16;
      { // build A = [h_r(64) | h_c(64) | radial | edge_attr | 0 pad] f16, stride 168
        int e = lane & 15, hh = lane >> 4;
        int eg = ebase + e;
        bool ok = eg < EPB;
        int r = ok ? er[eg] : 0, c = ok ? ec[eg] : 0;
        _Float16* arow = A + e * 168;
        const v4f* hr = (const v4f*)(hsh + r * 64);
        const v4f* hc = (const v4f*)(hsh + c * 64);
        if (hh == 0) {
#pragma unroll
          for (int k = 0; k < 16; ++k)
            *(v4h*)(arow + 4 * k) = __builtin_convertvector(ok ? hr[k] : vz4, v4h);
#pragma unroll
          for (int k = 0; k < 4; ++k)
            *(v4h*)(arow + 64 + 4 * k) = __builtin_convertvector(ok ? hc[k] : vz4, v4h);
        } else {
#pragma unroll
          for (int k = 4; k < 16; ++k)
            *(v4h*)(arow + 64 + 4 * k) = __builtin_convertvector(ok ? hc[k] : vz4, v4h);
          float rad = 0.0f;
          if (ok) {
#pragma unroll
            for (int d = 0; d < 3; ++d) { float df = crd[r*4+d] - crd[c*4+d]; rad += df * df; }
          }
          v4h tail = {};
          tail[0] = (_Float16)rad;
          tail[1] = ok ? (_Float16)eattr[eg] : (_Float16)0.0f;
          *(v4h*)(arow + 128) = tail;              // radial, eattr, 0, 0
          v4h zh = {};
#pragma unroll
          for (int k = 0; k < 7; ++k) *(v4h*)(arow + 132 + 4 * k) = zh;  // cols 132..159
        }
      }
      // GEMM1: [16x160] x [160x64] -> silu -> A f16 (cols 0..63)
      v8f m1[4] = {};
#pragma unroll
      for (int kt = 0; kt < 5; ++kt) {
        v16h a = frag_lds(A, 168, kt, lane);
#pragma unroll
        for (int nt = 0; nt < 4; ++nt)
          m1[nt] = wmma16(a, frag_glb(We1T, 160, nt, kt, lane), m1[nt]);
      }
#pragma unroll
      for (int nt = 0; nt < 4; ++nt) {
        float bb = be1[l*64 + nt*16 + ncol];
#pragma unroll
        for (int v = 0; v < 8; ++v)
          A[(mrow+v)*168 + nt*16 + ncol] = (_Float16)siluf(m1[nt][v] + bb);
      }
      // GEMM2: [16x64] x [64x64] -> silu -> F f32
      v8f m2[4] = {};
#pragma unroll
      for (int kt = 0; kt < 2; ++kt) {
        v16h a = frag_lds(A, 168, kt, lane);
#pragma unroll
        for (int nt = 0; nt < 4; ++nt)
          m2[nt] = wmma16(a, frag_glb(We2T, 64, nt, kt, lane), m2[nt]);
      }
#pragma unroll
      for (int nt = 0; nt < 4; ++nt) {
        float bb = be2[l*64 + nt*16 + ncol];
#pragma unroll
        for (int v = 0; v < 8; ++v)
          F[(mrow+v)*68 + nt*16 + ncol] = siluf(m2[nt][v] + bb);
      }
      // attention gate: m *= sigmoid(m @ Watt + batt); keep f32 in F, f16 in A
      if (lane < 16) {
        v4f* Fr = (v4f*)(F + lane * 68);
        const v4f* Wt = (const v4f*)(Watt + l * 64);
        v4f accv = {};
#pragma unroll
        for (int k = 0; k < 16; ++k) accv += Fr[k] * Wt[k];
        float g = sigmf(batt[l] + accv[0] + accv[1] + accv[2] + accv[3]);
#pragma unroll
        for (int k = 0; k < 16; ++k) {
          v4f v = Fr[k] * g;
          Fr[k] = v;
          *(v4h*)(A + lane * 168 + 4 * k) = __builtin_convertvector(v, v4h);
        }
      }
      // agg[r] += m  (LDS float atomics; two lanes per edge split the 64 cols)
      {
        int e = lane & 15, koff = (lane >> 4) * 32;
        int eg = ebase + e;
        if (eg < EPB) {
          int r = er[eg];
#pragma unroll
          for (int k = koff; k < koff + 32; k += 4) {
            v4f v = *(const v4f*)(F + e*68 + k);
            atomicAdd(&agg[r*64 + k + 0], v[0]);
            atomicAdd(&agg[r*64 + k + 1], v[1]);
            atomicAdd(&agg[r*64 + k + 2], v[2]);
            atomicAdd(&agg[r*64 + k + 3], v[3]);
          }
        }
      }
      // GEMM3: hidden = silu(m @ Wc1 + bc1) -> F
      v8f p3[4] = {};
#pragma unroll
      for (int kt = 0; kt < 2; ++kt) {
        v16h a = frag_lds(A, 168, kt, lane);
#pragma unroll
        for (int nt = 0; nt < 4; ++nt)
          p3[nt] = wmma16(a, frag_glb(Wc1T, 64, nt, kt, lane), p3[nt]);
      }
#pragma unroll
      for (int nt = 0; nt < 4; ++nt) {
        float bb = bc1[l*64 + nt*16 + ncol];
#pragma unroll
        for (int v = 0; v < 8; ++v)
          F[(mrow+v)*68 + nt*16 + ncol] = siluf(p3[nt][v] + bb);
      }
      // phi = hidden @ Wc2; trans = coord_diff * tanh(phi)*3; cacc[r] += trans
      if (lane < 16) {
        int eg = ebase + lane;
        if (eg < EPB) {
          const v4f* Fr = (const v4f*)(F + lane * 68);
          const v4f* Wp = (const v4f*)(Wc2 + l * 64);
          v4f accv = {};
#pragma unroll
          for (int k = 0; k < 16; ++k) accv += Fr[k] * Wp[k];
          float tf = tanhf(accv[0] + accv[1] + accv[2] + accv[3]) * 3.0f;
          int r = er[eg], c = ec[eg];
#pragma unroll
          for (int d = 0; d < 3; ++d) {
            float df = crd[r*4 + d] - crd[c*4 + d];
            atomicAdd(&cacc[r*4 + d], df * tf);
          }
        }
      }
    } // edge tiles
    __syncthreads();

    // ================= node MLP (waves 0,1) + coord update (tids 128..193) =========
    if (wave < 2) {
      const int base = wave * 16;
      { // A = [h | agg] f16, K=128 (rows >= 22 zero-padded)
        int e = lane & 15, hh = lane >> 4;
        int p = base + e;
        bool ok = p < NPART;
        _Float16* arow = A + e * 168;
        const v4f* src = (const v4f*)((hh == 0 ? hsh : agg) + (ok ? p : 0) * 64);
        _Float16* dst = arow + hh * 64;
#pragma unroll
        for (int k = 0; k < 16; ++k)
          *(v4h*)(dst + 4 * k) = __builtin_convertvector(ok ? src[k] : vz4, v4h);
      }
      v8f n1[4] = {};
#pragma unroll
      for (int kt = 0; kt < 4; ++kt) {
        v16h a = frag_lds(A, 168, kt, lane);
#pragma unroll
        for (int nt = 0; nt < 4; ++nt)
          n1[nt] = wmma16(a, frag_glb(Wn1T, 128, nt, kt, lane), n1[nt]);
      }
#pragma unroll
      for (int nt = 0; nt < 4; ++nt) {
        float bb = bn1[l*64 + nt*16 + ncol];
#pragma unroll
        for (int v = 0; v < 8; ++v)
          A[(mrow+v)*168 + nt*16 + ncol] = (_Float16)siluf(n1[nt][v] + bb);
      }
      v8f n2[4] = {};
#pragma unroll
      for (int kt = 0; kt < 2; ++kt) {
        v16h a = frag_lds(A, 168, kt, lane);
#pragma unroll
        for (int nt = 0; nt < 4; ++nt)
          n2[nt] = wmma16(a, frag_glb(Wn2T, 64, nt, kt, lane), n2[nt]);
      }
#pragma unroll
      for (int nt = 0; nt < 4; ++nt) {
        float bb = bn2[l*64 + nt*16 + ncol];
#pragma unroll
        for (int v = 0; v < 8; ++v) {
          int p = base + mrow + v;
          if (p < NPART) hsh[p*64 + nt*16 + ncol] += n2[nt][v] + bb;  // h = h + dh
        }
      }
    } else if (tid >= 128 && tid < 128 + NPART * 3) {
      int idx = tid - 128, p = idx / 3, d = idx - p * 3;
      crd[p*4 + d] += cacc[p*4 + d];
      cacc[p*4 + d] = 0.0f;
    }
    __syncthreads();
    for (int idx = tid; idx < NPART * HDIM; idx += 256) agg[idx] = 0.0f;
    __syncthreads();
  } // layers

  // vel = coord - x0, remove per-graph mean over particles
  if (tid < 3) {
    float s = 0.0f;
    for (int p = 0; p < NPART; ++p) s += crd[p*4 + tid] - crd0[p*4 + tid];
    meanv[tid] = s * (1.0f / (float)NPART);
  }
  __syncthreads();
  for (int idx = tid; idx < NPART * 3; idx += 256) {
    int p = idx / 3, d = idx - p * 3;
    out[b * (NPART * 3) + idx] = crd[p*4 + d] - crd0[p*4 + d] - meanv[d];
  }
}

extern "C" void kernel_launch(void* const* d_in, const int* in_sizes, int n_in,
                              void* d_out, int out_size, void* d_ws, size_t ws_size,
                              hipStream_t stream) {
  const float* t      = (const float*)d_in[0];
  const float* x      = (const float*)d_in[1];
  const float* h_init = (const float*)d_in[2];
  const int*   rows   = (const int*)d_in[3];
  const int*   cols   = (const int*)d_in[4];
  const float* Wemb   = (const float*)d_in[5];
  const float* bemb   = (const float*)d_in[6];
  const float* We1    = (const float*)d_in[7];
  const float* be1    = (const float*)d_in[8];
  const float* We2    = (const float*)d_in[9];
  const float* be2    = (const float*)d_in[10];
  const float* Watt   = (const float*)d_in[11];
  const float* batt   = (const float*)d_in[12];
  const float* Wn1    = (const float*)d_in[13];
  const float* bn1    = (const float*)d_in[14];
  const float* Wn2    = (const float*)d_in[15];
  const float* bn2    = (const float*)d_in[16];
  const float* Wc1    = (const float*)d_in[17];
  const float* bc1    = (const float*)d_in[18];
  const float* Wc2    = (const float*)d_in[19];
  _Float16* wsW = (_Float16*)d_ws;

  prep_weights<<<WS_HALVES / 256, 256, 0, stream>>>(We1, We2, Wc1, Wn1, Wn2, wsW);
  egnn_fused<<<1024, 256, 0, stream>>>(t, x, h_init, rows, cols, Wemb, bemb,
                                       be1, be2, Watt, batt, bn1, bn2, bc1, Wc2,
                                       wsW, (float*)d_out);
}
